// LMAGNN_LogicModel_78374563217885
// MI455X (gfx1250) — matrix operand order repaired and checked by reading, not compiled, not run on hardware
//
#include <hip/hip_runtime.h>
#include <hip/hip_bf16.h>

typedef __bf16 bf16_t;
typedef __attribute__((ext_vector_type(16))) __bf16 v16bf;
typedef __attribute__((ext_vector_type(8)))  float  v8f;

#define D   256
#define KT  8      // K tiles of 32 (D / 32)

// ---------------------------------------------------------------------------
// helpers
// ---------------------------------------------------------------------------
__device__ __forceinline__ void atomAddF(float* p, float v) {
  __hip_atomic_fetch_add(p, v, __ATOMIC_RELAXED, __HIP_MEMORY_SCOPE_AGENT);
}

// LDS index for the 16-bit WMMA A-fragment layout (16xK tile, K tiles of 32):
// lane L<16 holds K 0..7 (halves 0..7) and K 16..23 (halves 8..15) of row L;
// lane L+16 holds K 8..15 and K 24..31 of row L.
__device__ __forceinline__ int a_lds_index(int r, int k) {
  int kt = k >> 5, ko = k & 31, g = ko >> 3;
  int lane = r + ((g & 1) << 4);
  int idx  = ((g >> 1) << 3) | (ko & 7);
  return (kt * 32 + lane) * 16 + idx;
}

// ---------------------------------------------------------------------------
// utility kernels
// ---------------------------------------------------------------------------
__global__ void zero_f32(float* __restrict__ p, long long n) {
  long long i = (long long)blockIdx.x * blockDim.x + threadIdx.x;
  long long stride = (long long)gridDim.x * blockDim.x;
  for (; i < n; i += stride) p[i] = 0.f;
}

__global__ void fill_i32(int* __restrict__ p, int v, int n) {
  int i = blockIdx.x * blockDim.x + threadIdx.x;
  int stride = gridDim.x * blockDim.x;
  for (; i < n; i += stride) p[i] = v;
}

__global__ void pos_scatter(int* __restrict__ pos, const int* __restrict__ oi, int S) {
  int s = blockIdx.x * blockDim.x + threadIdx.x;
  if (s < S) pos[oi[s]] = s;
}

// Pack a [K,Nc] f32 matrix (optionally stored transposed as [Nc,K]) into the
// bf16 B-fragment layout: out[((nt*ktiles + kt)*32 + lane)*16 + j], where
// n = nt*16 + lane%16 ; k = kt*32 + (j<8 ? 0 : 16) + (lane<16 ? 0 : 8) + j%8.
__global__ void pack_b_kernel(const float* __restrict__ W, bf16_t* __restrict__ out,
                              int K, int Nc, int trans) {
  int idx = blockIdx.x * blockDim.x + threadIdx.x;
  if (idx >= K * Nc) return;
  int k = idx / Nc, n = idx % Nc;
  float v = trans ? W[(size_t)n * K + k] : W[(size_t)k * Nc + n];
  int nt = n >> 4, kt = k >> 5;
  int lane = (n & 15) + (((k >> 3) & 1) << 4);
  int j = (((k >> 4) & 1) << 3) | (k & 7);
  int ktiles = K >> 5;
  out[((size_t)(nt * ktiles + kt) * 32 + lane) * 16 + j] = (bf16_t)v;
}

// ---------------------------------------------------------------------------
// q_sem = (ent[subs] * rel[rels]) @ W_struct^T ; also emits ent[subs] rows
// ---------------------------------------------------------------------------
__global__ __launch_bounds__(D) void qsem_kernel(
    const int* __restrict__ subs, const int* __restrict__ rels,
    const float* __restrict__ ent, const float* __restrict__ rel,
    const float* __restrict__ W_struct, float* __restrict__ qsem,
    float* __restrict__ entsubs)
{
  __shared__ float tmp[D];
  const int b = blockIdx.x, t = threadIdx.x;
  const float* e = ent + (size_t)subs[b] * D;
  const float* r = rel + (size_t)rels[b] * D;
  float ev = e[t];
  entsubs[(size_t)b * D + t] = ev;
  tmp[t] = ev * r[t];
  __syncthreads();
  float s = 0.f;
  const float* wrow = W_struct + (size_t)t * D;   // q_sem[b][t] = sum_k tmp[k]*W[t][k]
  for (int k = 0; k < D; k++) s += tmp[k] * wrow[k];
  qsem[(size_t)b * D + t] = s;
}

// ---------------------------------------------------------------------------
// msg = relu((hidden[es] * rel[er]) @ Wmsg) scatter-added into agg[ed]
// block = 256 threads (8 waves) -> 64 edges (4 M-tiles) x 256 outputs.
// kt-outer loop: each wave loads its 2 B fragments once per K-step and runs
// 8 WMMAs against the 4 staged A tiles (4x less B traffic per edge).
// ---------------------------------------------------------------------------
__global__ __launch_bounds__(256) void msg_kernel(
    const float* __restrict__ hsrc, const float* __restrict__ rel,
    const int* __restrict__ es, const int* __restrict__ er, const int* __restrict__ ed,
    const bf16_t* __restrict__ Bp, float* __restrict__ agg)
{
  __shared__ __align__(32) bf16_t A[4][KT * 32 * 16];   // 4 x 8 KB
  __shared__ int eds[64];
  const int t = threadIdx.x;
  const int e0 = blockIdx.x * 64;
  const int r = t >> 4, c = t & 15;
  if (t < 64) eds[t] = ed[e0 + t];
  {
    const int kb = c * 16;
#pragma unroll
    for (int mt = 0; mt < 4; mt++) {
      const int e = e0 + mt * 16 + r;
      const float* __restrict__ h  = hsrc + (size_t)es[e] * D;
      const float* __restrict__ rr = rel  + (size_t)er[e] * D;
#pragma unroll
      for (int jj = 0; jj < 16; jj++) {
        int k = kb + jj;
        A[mt][a_lds_index(r, k)] = (bf16_t)(h[k] * rr[k]);
      }
    }
  }
  __syncthreads();
  const int w = t >> 5, lane = t & 31;
  const int nt0 = w * 2, nt1 = nt0 + 1;
  v8f acc[4][2];
#pragma unroll
  for (int mt = 0; mt < 4; mt++) { acc[mt][0] = v8f{}; acc[mt][1] = v8f{}; }
#pragma unroll
  for (int kt = 0; kt < KT; kt++) {
    v16bf b0 = *(const v16bf*)&Bp[((size_t)(nt0 * KT + kt) * 32 + lane) * 16];
    v16bf b1 = *(const v16bf*)&Bp[((size_t)(nt1 * KT + kt) * 32 + lane) * 16];
#pragma unroll
    for (int mt = 0; mt < 4; mt++) {
      v16bf a = *(const v16bf*)&A[mt][(kt * 32 + lane) * 16];
      acc[mt][0] = __builtin_amdgcn_wmma_f32_16x16x32_bf16(false, a, false, b0, (short)0, acc[mt][0], false, false);
      acc[mt][1] = __builtin_amdgcn_wmma_f32_16x16x32_bf16(false, a, false, b1, (short)0, acc[mt][1], false, false);
    }
  }
  const int n0 = nt0 * 16 + (lane & 15);
  const int n1 = nt1 * 16 + (lane & 15);
  const int mb = (lane >> 4) << 3;
#pragma unroll
  for (int mt = 0; mt < 4; mt++) {
#pragma unroll
    for (int v = 0; v < 8; v++) {
      int m = mt * 16 + mb + v;
      float* row = agg + (size_t)eds[m] * D;
      float x0 = acc[mt][0][v], x1 = acc[mt][1][v];
      atomAddF(row + n0, x0 > 0.f ? x0 : 0.f);
      atomAddF(row + n1, x1 > 0.f ? x1 : 0.f);
    }
  }
}

// ---------------------------------------------------------------------------
// GRU step for 16 rows: gi = agg@Wih^T, gh = h@Whh^T (bf16 WMMA), then gating.
// h row j = (pos[si[j]] >= 0) ? hprev[pos[si[j]]] : 0
// ---------------------------------------------------------------------------
__global__ __launch_bounds__(256) void gru_kernel(
    const float* __restrict__ agg, const float* __restrict__ hprev,
    const int* __restrict__ pos, const int* __restrict__ si,
    const bf16_t* __restrict__ Bih, const bf16_t* __restrict__ Bhh,
    const float* __restrict__ b_ih, const float* __restrict__ b_hh,
    float* __restrict__ hout)
{
  __shared__ __align__(32) bf16_t Ax[KT * 32 * 16];   // 8 KB
  __shared__ __align__(32) bf16_t Ah[KT * 32 * 16];   // 8 KB
  __shared__ float Hf[16 * D];                        // 16 KB
  __shared__ float Arz[16 * 512];                     // 32 KB (r | z pre-acts, gi+gh)
  __shared__ float Cn[16 * D];                        // 16 KB (gi n-channel)
  __shared__ float Hn[16 * D];                        // 16 KB (gh n-channel)
  const int t = threadIdx.x;
  const int j0 = blockIdx.x * 16;
  const int r = t >> 4, c = t & 15;
  {
    const int j = j0 + r;
    const int p = pos[si[j]];
    const float* __restrict__ hr = (p >= 0) ? (hprev + (size_t)p * D) : nullptr;
    const float* __restrict__ xr = agg + (size_t)j * D;
    const int kb = c * 16;
#pragma unroll
    for (int jj = 0; jj < 16; jj++) {
      int k = kb + jj;
      float hv = hr ? hr[k] : 0.f;
      Ax[a_lds_index(r, k)] = (bf16_t)xr[k];
      Ah[a_lds_index(r, k)] = (bf16_t)hv;
      Hf[r * D + k] = hv;
    }
  }
  __syncthreads();
  const int w = t >> 5, lane = t & 31;
  const int mb = (lane >> 4) << 3;
  const int nl = lane & 15;
  for (int q = 0; q < 6; q++) {            // 48 N-tiles (768 cols) over 8 waves
    const int nt = w * 6 + q;
    v8f ai = {}, ah = {};
#pragma unroll
    for (int kt = 0; kt < KT; kt++) {
      v16bf ax  = *(const v16bf*)&Ax[(kt * 32 + lane) * 16];
      v16bf axh = *(const v16bf*)&Ah[(kt * 32 + lane) * 16];
      v16bf bi  = *(const v16bf*)&Bih[((size_t)(nt * KT + kt) * 32 + lane) * 16];
      v16bf bh  = *(const v16bf*)&Bhh[((size_t)(nt * KT + kt) * 32 + lane) * 16];
      ai = __builtin_amdgcn_wmma_f32_16x16x32_bf16(false, ax,  false, bi, (short)0, ai, false, false);
      ah = __builtin_amdgcn_wmma_f32_16x16x32_bf16(false, axh, false, bh, (short)0, ah, false, false);
    }
    const int ch = nt >> 4;                // 0 = r-gate cols, 1 = z, 2 = n
    const int dl = (nt & 15) * 16 + nl;    // 0..255
#pragma unroll
    for (int v = 0; v < 8; v++) {
      int m = mb + v;
      if (ch == 0)      Arz[m * 512 + dl]       = ai[v] + ah[v];
      else if (ch == 1) Arz[m * 512 + 256 + dl] = ai[v] + ah[v];
      else { Cn[m * D + dl] = ai[v]; Hn[m * D + dl] = ah[v]; }
    }
  }
  __syncthreads();
  {
    const int kb = c * 16;
#pragma unroll
    for (int jj = 0; jj < 16; jj++) {
      int d = kb + jj;
      float Apre = Arz[r * 512 + d]       + b_ih[d]       + b_hh[d];
      float Zpre = Arz[r * 512 + 256 + d] + b_ih[256 + d] + b_hh[256 + d];
      float rg = 1.f / (1.f + expf(-Apre));
      float zg = 1.f / (1.f + expf(-Zpre));
      float nn = tanhf(Cn[r * D + d] + b_ih[512 + d] + rg * (Hn[r * D + d] + b_hh[512 + d]));
      hout[(size_t)(j0 + r) * D + d] = (1.f - zg) * nn + zg * Hf[r * D + d];
    }
  }
}

// ---------------------------------------------------------------------------
// per-hop scoring: scores[nb,ne] += gamma_norm * (feats@W_logic + <q_sem[nb], ent[ne]>)
// one wave per node
// ---------------------------------------------------------------------------
__global__ __launch_bounds__(256) void score_kernel(
    const float* __restrict__ feats, const int* __restrict__ nb, const int* __restrict__ ne,
    const float* __restrict__ qsem, const float* __restrict__ ent,
    const float* __restrict__ W_logic, const float* __restrict__ gamma,
    float* __restrict__ scores, int Ni, int NENT, int hop)
{
  const int w = threadIdx.x >> 5, lane = threadIdx.x & 31;
  const int j = blockIdx.x * 8 + w;
  if (j >= Ni) return;
  float g0 = gamma[0], g1 = gamma[1], g2 = gamma[2];
  float mx = fmaxf(g0, fmaxf(g1, g2));
  float e0 = expf(g0 - mx), e1 = expf(g1 - mx), e2 = expf(g2 - mx);
  float sel = (hop == 0) ? e0 : ((hop == 1) ? e1 : e2);
  float gn = sel / (e0 + e1 + e2);
  const int b = nb[j], n = ne[j];
  const float* f  = feats + (size_t)j * D;
  const float* q  = qsem  + (size_t)b * D;
  const float* en = ent   + (size_t)n * D;
  float s = 0.f;
#pragma unroll
  for (int k = lane; k < D; k += 32) s += f[k] * W_logic[k] + q[k] * en[k];
  for (int off = 16; off > 0; off >>= 1) s += __shfl_down(s, off, 32);
  if (lane == 0) atomAddF(&scores[(size_t)b * NENT + n], gn * s);
}

// ---------------------------------------------------------------------------
// host orchestration
// ---------------------------------------------------------------------------
extern "C" void kernel_launch(void* const* d_in, const int* in_sizes, int n_in,
                              void* d_out, int out_size, void* d_ws, size_t ws_size,
                              hipStream_t stream) {
  (void)n_in; (void)ws_size;
  const int* subs = (const int*)d_in[0];
  const int* rels = (const int*)d_in[1];
  const int *nb[3], *ne[3], *es[3], *er[3], *ed[3], *oi[3], *si[3];
  int N[3], E[3], S[3];
  for (int i = 0; i < 3; i++) {
    int b = 2 + i * 7;
    nb[i] = (const int*)d_in[b];     ne[i] = (const int*)d_in[b + 1];
    es[i] = (const int*)d_in[b + 2]; er[i] = (const int*)d_in[b + 3];
    ed[i] = (const int*)d_in[b + 4]; oi[i] = (const int*)d_in[b + 5];
    si[i] = (const int*)d_in[b + 6];
    N[i] = in_sizes[b]; E[i] = in_sizes[b + 2]; S[i] = in_sizes[b + 5];
  }
  const int Mh[3] = {16384, 65536, 131072};
  const float* ent      = (const float*)d_in[23];
  const float* rel      = (const float*)d_in[24];
  const float* gamma    = (const float*)d_in[25];
  const float* W_logic  = (const float*)d_in[26];
  const float* W_struct = (const float*)d_in[27];
  const float* W_msg    = (const float*)d_in[28];
  const float* W_ih     = (const float*)d_in[29];
  const float* W_hh     = (const float*)d_in[30];
  const float* b_ih     = (const float*)d_in[31];
  const float* b_hh     = (const float*)d_in[32];
  const int NENT = in_sizes[23] / D;
  const int B = in_sizes[0];
  float* scores = (float*)d_out;

  // workspace carve-up
  char* ws = (char*)d_ws;
  size_t off = 0;
  auto carve = [&](size_t bytes) -> void* {
    void* p = ws + off; off += (bytes + 255) & ~(size_t)255; return p;
  };
  float* buf0    = (float*)carve((size_t)N[2] * D * 4);
  float* buf1    = (float*)carve((size_t)N[2] * D * 4);
  float* agg     = (float*)carve((size_t)N[2] * D * 4);
  float* entsubs = (float*)carve((size_t)B * D * 4);
  float* h0      = (float*)carve((size_t)B * D * 4);
  float* qsem    = (float*)carve((size_t)B * D * 4);
  int*   pos     = (int*)  carve((size_t)Mh[2] * 4);
  bf16_t* Wmsg_p = (bf16_t*)carve((size_t)3 * D * D * 2);
  bf16_t* Wih_p  = (bf16_t*)carve((size_t)3 * D * D * 2);
  bf16_t* Whh_p  = (bf16_t*)carve((size_t)3 * D * D * 2);

  // 1) zero output + h0
  zero_f32<<<2048, 256, 0, stream>>>(scores, (long long)out_size);
  zero_f32<<<64, 256, 0, stream>>>(h0, (long long)B * D);

  // 2) q_sem + ent[subs]
  qsem_kernel<<<B, D, 0, stream>>>(subs, rels, ent, rel, W_struct, qsem, entsubs);

  // 3) pack weights to bf16 WMMA fragment layout
  for (int i = 0; i < 3; i++)
    pack_b_kernel<<<(D * D + 255) / 256, 256, 0, stream>>>(
        W_msg + (size_t)i * D * D, Wmsg_p + (size_t)i * D * D, D, D, 0);
  pack_b_kernel<<<(D * 3 * D + 255) / 256, 256, 0, stream>>>(W_ih, Wih_p, D, 3 * D, 1);
  pack_b_kernel<<<(D * 3 * D + 255) / 256, 256, 0, stream>>>(W_hh, Whh_p, D, 3 * D, 1);

  // 4) hop loop
  for (int i = 0; i < 3; i++) {
    const float* msg_src = (i == 0) ? entsubs : ((i == 1) ? buf0 : buf1);
    const float* gru_src = (i == 0) ? h0      : ((i == 1) ? buf0 : buf1);
    float* hout = (i & 1) ? buf1 : buf0;

    zero_f32<<<2048, 256, 0, stream>>>(agg, (long long)N[i] * D);
    msg_kernel<<<E[i] / 64, 256, 0, stream>>>(
        msg_src, rel, es[i], er[i], ed[i], Wmsg_p + (size_t)i * D * D, agg);

    fill_i32<<<(Mh[i] + 255) / 256, 256, 0, stream>>>(pos, -1, Mh[i]);
    pos_scatter<<<(S[i] + 255) / 256, 256, 0, stream>>>(pos, oi[i], S[i]);

    gru_kernel<<<N[i] / 16, 256, 0, stream>>>(
        agg, gru_src, pos, si[i], Wih_p, Whh_p, b_ih, b_hh, hout);

    score_kernel<<<(N[i] + 7) / 8, 256, 0, stream>>>(
        hout, nb[i], ne[i], qsem, ent, W_logic, gamma, scores, N[i], NENT, i);
  }
}
